// FitTorch_58102317580440
// MI455X (gfx1250) — matrix operand order repaired and checked by reading, not compiled, not run on hardware
//
#include <hip/hip_runtime.h>
#include <hip/hip_bf16.h>

#define NATOMS  50000
#define ND      64
#define NSTRUCT 250
#define NPAIRS  1000000
#define H1D     128
#define H2D     64
#define NTILES  (NATOMS / 16)   // 3125, exact
#define WPB     2               // waves per block

typedef __attribute__((ext_vector_type(16))) _Float16 v16h;
typedef __attribute__((ext_vector_type(8)))  float    v8f;
typedef __attribute__((ext_vector_type(2)))  float    f2;

#if defined(__has_builtin)
#if __has_builtin(__builtin_amdgcn_tanhf)
#define FAST_TANH(x) __builtin_amdgcn_tanhf(x)
#endif
#endif
#ifndef FAST_TANH
#define FAST_TANH(x) tanhf(x)
#endif

__device__ __forceinline__ void lds_fence() {
    asm volatile("s_wait_dscnt 0" ::: "memory");
}

__device__ __forceinline__ v8f wmma_f16(v16h a, v16h b, v8f c) {
    return __builtin_amdgcn_wmma_f32_16x16x32_f16(false, a, false, b, (short)0, c,
                                                  false, false);
}

// A-fragment (16xK f16, ISA 7.12.2) packed-LDS index for element (m, k):
// frag ks = k>>5; lane = m + 16*kh; within-lane t = 2*i + parity.
__device__ __forceinline__ int packA_idx(int m, int k) {
    int kl = k & 31;
    int kh = (kl >> 3) & 1;                      // K in [8,16)u[24,32) -> hi lanes
    int i  = ((kl & 16) >> 2) + ((kl >> 1) & 3); // vgpr index 0..7
    int t  = 2 * i + (kl & 1);
    return (((k >> 5) * 32) + m + 16 * kh) * 16 + t;
}

// ---------------------------------------------------------------------------
__global__ void zero_out_kernel(float* __restrict__ out, int n) {
    int i = blockIdx.x * blockDim.x + threadIdx.x;
    if (i < n) out[i] = 0.0f;
}

// ---------------------------------------------------------------------------
// Fused MLP fwd+bwd, persistent waves, fragment-packed LDS everywhere.
__global__ __launch_bounds__(32 * WPB)
void mlp_fused_kernel(const float* __restrict__ x,
                      const int*   __restrict__ indices,
                      const float* __restrict__ W1, const float* __restrict__ b1,
                      const float* __restrict__ W2, const float* __restrict__ b2,
                      const float* __restrict__ W3, const float* __restrict__ b3,
                      float* __restrict__ energy,    // d_out[0:NSTRUCT]
                      float* __restrict__ dEdD)      // ws: NATOMS x ND
{
    // B-fragment-packed weights: frag f -> 32 lanes x 16 contiguous halfs.
    __shared__ alignas(32) _Float16 P1f[16 * 512];   // W1   : ks 2, nt 8
    __shared__ alignas(32) _Float16 P2f[16 * 512];   // W2   : ks 4, nt 4
    __shared__ alignas(32) _Float16 P2t[16 * 512];   // W2^T : ks 2, nt 8
    __shared__ alignas(32) _Float16 P1t[16 * 512];   // W1^T : ks 4, nt 4
    __shared__ float sB1[H1D], sB2[H2D], sW3[H2D];
    // A-fragment-packed activations, per wave.
    __shared__ alignas(32) _Float16 pH1 [WPB][4 * 512];  // 16 x 128
    __shared__ alignas(32) _Float16 pDZ2[WPB][2 * 512];  // 16 x 64
    __shared__ alignas(32) _Float16 pDZ1[WPB][4 * 512];  // 16 x 128

    const int tid = threadIdx.x;

    // ---- stage packed weights (B layout: K_local = (t&8)*2 + (lane>>4)*8 + (t&7)) ----
    for (int e = tid; e < 8192; e += blockDim.x) {
        int f = e >> 9, ln = (e >> 4) & 31, t = e & 15;
        int kl = ((t & 8) << 1) + ((ln >> 4) << 3) + (t & 7), nl = ln & 15;
        { int ks = f & 1, nt = f >> 1;          // P1f: B[k][n] = W1[k*H1D+n]
          P1f[e] = (_Float16)W1[(ks * 32 + kl) * H1D + nt * 16 + nl]; }
        { int ks = f & 3, nt = f >> 2;          // P2f: B[k][n] = W2[k*H2D+n]
          P2f[e] = (_Float16)W2[(ks * 32 + kl) * H2D + nt * 16 + nl]; }
        { int ks = f & 1, nt = f >> 1;          // P2t: B[k][n] = W2[n*H2D+k]
          P2t[e] = (_Float16)W2[(nt * 16 + nl) * H2D + ks * 32 + kl]; }
        { int ks = f & 3, nt = f >> 2;          // P1t: B[k][n] = W1[n*H1D+k]
          P1t[e] = (_Float16)W1[(nt * 16 + nl) * H1D + ks * 32 + kl]; }
    }
    for (int i = tid; i < H1D; i += blockDim.x) sB1[i] = b1[i];
    for (int i = tid; i < H2D; i += blockDim.x) { sB2[i] = b2[i]; sW3[i] = W3[i]; }
    __syncthreads();

    const int lane  = tid & 31;
    const int wave  = tid >> 5;
    const int n16   = lane & 15;
    const int mbase = (lane >> 4) * 8;
    const int kh8   = ((lane >> 4) & 1) * 8;
    const float b3v = b3[0];
    const int nwaves = gridDim.x * WPB;

    for (int tile = blockIdx.x * WPB + wave; tile < NTILES; tile += nwaves) {
        const int atom0 = tile * 16;

        // ---- x A-fragments straight from global (fp32 -> f16) ----
        v16h ax[2];
        {
            const float* xr = x + (size_t)(atom0 + n16) * ND;
#pragma unroll
            for (int ks = 0; ks < 2; ++ks)
#pragma unroll
                for (int i = 0; i < 8; ++i) {
                    int kb = ks * 32 + ((i < 4) ? 0 : 16) + kh8 + (i & 3) * 2;
                    f2 v = *(const f2*)(xr + kb);
                    ax[ks][2 * i]     = (_Float16)v.x;
                    ax[ks][2 * i + 1] = (_Float16)v.y;
                }
        }

        // ---- GEMM1: h1 = tanh(x@W1 + b1) -> pH1 (A-packed) ----
#pragma unroll
        for (int nt = 0; nt < 8; ++nt) {
            v8f acc = {};
#pragma unroll
            for (int ks = 0; ks < 2; ++ks) {
                v16h b = *(const v16h*)&P1f[((nt * 2 + ks) << 9) + (lane << 4)];
                acc = wmma_f16(ax[ks], b, acc);
            }
            int   n    = nt * 16 + n16;
            float bias = sB1[n];
#pragma unroll
            for (int r = 0; r < 8; ++r) {
                float h = FAST_TANH(acc[r] + bias);
                pH1[wave][packA_idx(mbase + r, n)] = (_Float16)h;
            }
        }
        lds_fence();

        // ---- GEMM2: h2 = tanh(h1@W2 + b2); e += h2.W3; dz2 -> pDZ2 ----
        v16h a2[4];
#pragma unroll
        for (int ks = 0; ks < 4; ++ks)
            a2[ks] = *(const v16h*)&pH1[wave][(ks << 9) + (lane << 4)];
        float er[8] = {0, 0, 0, 0, 0, 0, 0, 0};
#pragma unroll
        for (int nt = 0; nt < 4; ++nt) {
            v8f acc = {};
#pragma unroll
            for (int ks = 0; ks < 4; ++ks) {
                v16h b = *(const v16h*)&P2f[((nt * 4 + ks) << 9) + (lane << 4)];
                acc = wmma_f16(a2[ks], b, acc);
            }
            int   n    = nt * 16 + n16;
            float bias = sB2[n];
            float w3   = sW3[n];
#pragma unroll
            for (int r = 0; r < 8; ++r) {
                float h = FAST_TANH(acc[r] + bias);
                er[r] += h * w3;
                pDZ2[wave][packA_idx(mbase + r, n)] =
                    (_Float16)((1.0f - h * h) * w3);
            }
        }

        // ---- per-atom energy: reduce er over 16 lanes of each half ----
#pragma unroll
        for (int r = 0; r < 8; ++r) {
            float e = er[r];
            e += __shfl_xor(e, 1);
            e += __shfl_xor(e, 2);
            e += __shfl_xor(e, 4);
            e += __shfl_xor(e, 8);
            if (n16 == 0)
                atomicAdd(&energy[indices[atom0 + mbase + r]], e + b3v);
        }
        lds_fence();

        // ---- bwd GEMM1: dh1 = dz2 @ W2^T ; dz1 = dh1*(1-h1^2) -> pDZ1 ----
        v16h ad2[2];
#pragma unroll
        for (int ks = 0; ks < 2; ++ks)
            ad2[ks] = *(const v16h*)&pDZ2[wave][(ks << 9) + (lane << 4)];
#pragma unroll
        for (int nt = 0; nt < 8; ++nt) {
            v8f acc = {};
#pragma unroll
            for (int ks = 0; ks < 2; ++ks) {
                v16h b = *(const v16h*)&P2t[((nt * 2 + ks) << 9) + (lane << 4)];
                acc = wmma_f16(ad2[ks], b, acc);
            }
            int n = nt * 16 + n16;
#pragma unroll
            for (int r = 0; r < 8; ++r) {
                int   ia = packA_idx(mbase + r, n);
                float h  = (float)pH1[wave][ia];
                pDZ1[wave][ia] = (_Float16)(acc[r] * (1.0f - h * h));
            }
        }
        lds_fence();

        // ---- bwd GEMM2: dEdD = dz1 @ W1^T -> global ws ----
        v16h ad1[4];
#pragma unroll
        for (int ks = 0; ks < 4; ++ks)
            ad1[ks] = *(const v16h*)&pDZ1[wave][(ks << 9) + (lane << 4)];
#pragma unroll
        for (int nt = 0; nt < 4; ++nt) {
            v8f acc = {};
#pragma unroll
            for (int ks = 0; ks < 4; ++ks) {
                v16h b = *(const v16h*)&P1t[((nt * 4 + ks) << 9) + (lane << 4)];
                acc = wmma_f16(ad1[ks], b, acc);
            }
            int n = nt * 16 + n16;
#pragma unroll
            for (int r = 0; r < 8; ++r)
                dEdD[(size_t)(atom0 + mbase + r) * ND + n] = acc[r];
        }
        lds_fence();   // WAR guard before next tile rewrites pH1/pDZ*
    }
}

// ---------------------------------------------------------------------------
// One wave per pair-triple; xd is a 768 MB one-shot stream -> non-temporal,
// dEdD (12.8 MB) stays L2-resident for the gathers.
__global__ __launch_bounds__(256)
void pair_force_kernel(const float* __restrict__ xd,
                       const int*   __restrict__ xd_indx,
                       const int*   __restrict__ unique_j,
                       const float* __restrict__ dEdD,
                       float* __restrict__ forces)   // d_out + NSTRUCT
{
    const int lane   = threadIdx.x & 31;
    const int wave   = (blockIdx.x * blockDim.x + threadIdx.x) >> 5;
    const int nwaves = (gridDim.x * blockDim.x) >> 5;

    for (int p = wave; p < NPAIRS; p += nwaves) {
        const int neigh = xd_indx[6 * p];                 // xd_indx[3p][0]
        const f2  gv = *(const f2*)(dEdD + (size_t)neigh * ND + 2 * lane);

        f2 xv[3];
#pragma unroll
        for (int c = 0; c < 3; ++c)
            xv[c] = __builtin_nontemporal_load(
                (const f2*)(xd + (size_t)(3 * p + c) * ND + 2 * lane));

#pragma unroll
        for (int c = 0; c < 3; ++c) {
            float s = xv[c].x * gv.x + xv[c].y * gv.y;
            s += __shfl_down(s, 16);
            s += __shfl_down(s, 8);
            s += __shfl_down(s, 4);
            s += __shfl_down(s, 2);
            s += __shfl_down(s, 1);
            if (lane == 0) {
                int j = unique_j[3 * p + c];
                atomicAdd(&forces[3 * j + c], -s);
            }
        }
    }
}

// ---------------------------------------------------------------------------
extern "C" void kernel_launch(void* const* d_in, const int* in_sizes, int n_in,
                              void* d_out, int out_size, void* d_ws, size_t ws_size,
                              hipStream_t stream) {
    const float* x        = (const float*)d_in[0];
    const float* xd       = (const float*)d_in[1];
    const int*   indices  = (const int*)  d_in[2];
    const int*   xd_indx  = (const int*)  d_in[4];
    const int*   unique_j = (const int*)  d_in[5];
    const float* W1       = (const float*)d_in[6];
    const float* b1       = (const float*)d_in[7];
    const float* W2       = (const float*)d_in[8];
    const float* b2       = (const float*)d_in[9];
    const float* W3       = (const float*)d_in[10];
    const float* b3       = (const float*)d_in[11];

    float* out    = (float*)d_out;           // [0:250] energy, [250:] forces
    float* dEdD   = (float*)d_ws;            // NATOMS x ND floats
    float* forces = out + NSTRUCT;

    zero_out_kernel<<<(out_size + 255) / 256, 256, 0, stream>>>(out, out_size);

    mlp_fused_kernel<<<256, 32 * WPB, 0, stream>>>(
        x, indices, W1, b1, W2, b2, W3, b3, out, dEdD);

    pair_force_kernel<<<4096, 256, 0, stream>>>(xd, xd_indx, unique_j, dEdD, forces);
}
